// InLayer_25486335935239
// MI455X (gfx1250) — matrix operand order repaired and checked by reading, not compiled, not run on hardware
//
#include <hip/hip_runtime.h>
#include <hip/hip_bf16.h>

typedef __attribute__((ext_vector_type(16))) __bf16       v16bf;
typedef __attribute__((ext_vector_type(2)))  __bf16       bf16x2;
typedef __attribute__((ext_vector_type(8)))  float        v8f;
typedef __attribute__((ext_vector_type(8)))  unsigned int v8u;

#define N_OBJ 60
#define P_F   16
#define DE_F  16
#define DO_F  16
#define NR_E  3540
#define NTE   222      // ceil(3540/16)
#define NWAVE 8

__device__ __forceinline__ v8f wmma_bf16(v16bf a, v16bf b, v8f c) {
  return __builtin_amdgcn_wmma_f32_16x16x32_bf16(false, a, false, b, (short)0, c, false, false);
}

__device__ __forceinline__ float reluf(float a) { return a > 0.f ? a : 0.f; }

// convert pair to bf16 (v_cvt_pk_bf16_f32) then packed ReLU (v_pk_max_num_bf16).
// cvt output is canonical -> no canonicalize needed; round-then-clamp == clamp-then-round.
__device__ __forceinline__ unsigned int pack2relu(float a, float b) {
  bf16x2 t;
  t[0] = (__bf16)a;
  t[1] = (__bf16)b;
  bf16x2 z = {(__bf16)0.0f, (__bf16)0.0f};
  t = __builtin_elementwise_max(t, z);
  return __builtin_bit_cast(unsigned int, t);
}

__device__ __forceinline__ unsigned int pack2(float a, float b) {
  bf16x2 t;
  t[0] = (__bf16)a;
  t[1] = (__bf16)b;
  return __builtin_bit_cast(unsigned int, t);
}

// lane <-> lane^16 exchange as pure VALU: v_permlanex16_b32, identity selects
__device__ __forceinline__ unsigned int xhalf(unsigned int v) {
  return __builtin_amdgcn_permlanex16(v, v, 0x76543210u, 0xfedcba98u, false, false);
}

// Weight fragment as WMMA *A* operand: row m = neuron, k order (permK: ISA 16-bit A table):
// k = ko + ((v&4)?16:0) + half*8 + 2*(v&3); else contiguous k = ko + half*16 + 2*v.
// permK: consumer-side feature permutation phi(k) = (k&16) ? k^8 : k, which
// absorbs the half-swap of the in-register transpose (see next_bfrags).
__device__ __forceinline__ v16bf load_w_afrag(const float* __restrict__ W, int din,
                                              int neuron, int ko, int half, bool permK) {
  v16bf f;
#pragma unroll
  for (int v = 0; v < 8; ++v) {
    int k = permK ? (ko + ((v & 4) ? 16 : 0) + half * 8 + 2 * (v & 3)) : (ko + half * 16 + 2 * v);
    if (permK) k = (k & 16) ? (k ^ 8) : k;
    f[2 * v]     = (__bf16)W[neuron * din + k];
    f[2 * v + 1] = (__bf16)W[neuron * din + k + 1];
  }
  return f;
}

// Bias as WMMA C operand: C[m][n] = bias[m];  c[v] = bias[base + 8*half + v]
__device__ __forceinline__ v8f load_bias_c(const float* __restrict__ bptr, int base, int half) {
  v8f c;
#pragma unroll
  for (int v = 0; v < 8; ++v) c[v] = bptr[base + 8 * half + v];
  return c;
}

// acc[q] holds features (q^half)*16 + 8*half + v (edge column = lane%16).
// With the phi column permutation in the consumer's weight fragments, the
// next-layer B fragments are position-fixed for both lane halves:
//   frag f:  low dwords = pk[2f],  high dwords = xhalf(pk[2f+1])  -- no selects.
__device__ __forceinline__ void next_bfrags(const v8f* acc, v16bf& f0, v16bf& f1) {
  unsigned int pk[4][4];
#pragma unroll
  for (int ct = 0; ct < 4; ++ct) {
#pragma unroll
    for (int j = 0; j < 4; ++j) {
      pk[ct][j] = pack2relu(acc[ct][2 * j], acc[ct][2 * j + 1]);
    }
  }
  v8u u0, u1;
#pragma unroll
  for (int j = 0; j < 4; ++j) {
    u0[j]     = pk[0][j];
    u0[4 + j] = xhalf(pk[1][j]);
    u1[j]     = pk[2][j];
    u1[4 + j] = xhalf(pk[3][j]);
  }
  f0 = __builtin_bit_cast(v16bf, u0);
  f1 = __builtin_bit_cast(v16bf, u1);
}

__global__ __launch_bounds__(256, 1)
void in_fused_kernel(const float* __restrict__ x,
                     const float* __restrict__ fr1w, const float* __restrict__ fr1b,
                     const float* __restrict__ fr2w, const float* __restrict__ fr2b,
                     const float* __restrict__ fr3w, const float* __restrict__ fr3b,
                     const float* __restrict__ fo1w, const float* __restrict__ fo1b,
                     const float* __restrict__ fo2w, const float* __restrict__ fo2b,
                     const float* __restrict__ fo3w, const float* __restrict__ fo3b,
                     float* __restrict__ out)
{
  __shared__ __attribute__((aligned(32))) __bf16 xbf[N_OBJ * P_F]; // x[b] node-major
  __shared__ __attribute__((aligned(16))) float  ebar[N_OBJ * DE_F];

  const int tid  = threadIdx.x;
  const int lane = tid & 31;
  const int wave = tid >> 5;
  const int half = lane >> 4;
  const int lrow = lane & 15;
  const int b    = blockIdx.x;

  // ---- phase 0: stage x[b] into LDS (node-major bf16), zero ebar ----
  for (int i = tid; i < N_OBJ * P_F; i += blockDim.x) {
    int n = i >> 4, p = i & 15;
    xbf[n * P_F + p] = (__bf16)x[(size_t)b * (P_F * N_OBJ) + p * N_OBJ + n];
    ebar[i] = 0.0f;
  }
  __syncthreads();

  // ---- persistent weight A-fragments + bias C-vectors (relational MLP) ----
  // accumulator q maps to neuron block (q ^ half): permuted row order so the
  // transpose exchange needs no lane-dependent selects.
  v16bf W1[4], W2[4][2], W3[2];
  v8f   C1b[4], C2b[4], C3b;
#pragma unroll
  for (int ct = 0; ct < 4; ++ct) {
    const int cte = ct ^ half;
    W1[ct]    = load_w_afrag(fr1w, 32, cte * 16 + lrow, 0,  half, false);
    W2[ct][0] = load_w_afrag(fr2w, 64, cte * 16 + lrow, 0,  half, true);
    W2[ct][1] = load_w_afrag(fr2w, 64, cte * 16 + lrow, 32, half, true);
    C1b[ct]   = load_bias_c(fr1b, cte * 16, half);
    C2b[ct]   = load_bias_c(fr2b, cte * 16, half);
  }
  W3[0] = load_w_afrag(fr3w, 64, lrow, 0,  half, true);
  W3[1] = load_w_afrag(fr3w, 64, lrow, 32, half, true);
  C3b   = load_bias_c(fr3b, 0, half);

  // ---- phase 1: relational MLP, 16 edge-columns per tile per wave ----
  for (int t = wave; t < NTE; t += NWAVE) {
    const int e  = t * 16 + lrow;
    const int ec = e < NR_E ? e : (NR_E - 1);        // clamp: no predicated loads
    const int r  = ec / 59;
    const int jj = ec - r * 59;
    const int s  = jj + (jj >= r ? 1 : 0);
    const int node = half ? s : r;                   // h=0: K 0..15 (recv), h=1: 16..31 (send)

    v8u bu = *(const v8u*)&xbf[node * P_F];          // 32B aligned -> 2x ds_load_b128
    v16bf B1f = __builtin_bit_cast(v16bf, bu);

    // layer 1: D = W1 x B + bias   (D: lane = edge col, v = feature)
    v8f acc[4];
#pragma unroll
    for (int ct = 0; ct < 4; ++ct) acc[ct] = wmma_bf16(W1[ct], B1f, C1b[ct]);

    // layer 2: 64 -> 64
    v16bf B2f0, B2f1;
    next_bfrags(acc, B2f0, B2f1);
#pragma unroll
    for (int ct = 0; ct < 4; ++ct) {
      acc[ct] = wmma_bf16(W2[ct][0], B2f0, C2b[ct]);
      acc[ct] = wmma_bf16(W2[ct][1], B2f1, acc[ct]);
    }

    // layer 3: 64 -> 16 (edge embedding)
    v16bf B3f0, B3f1;
    next_bfrags(acc, B3f0, B3f1);
    v8f e3 = wmma_bf16(W3[0], B3f0, C3b);
    e3     = wmma_bf16(W3[1], B3f1, e3);

    // receiver scatter-add: lane owns one edge column; de = v + 8*half
    if (e < NR_E) {
#pragma unroll
      for (int v = 0; v < 8; ++v) {
        atomicAdd(&ebar[r * DE_F + 8 * half + v], reluf(e3[v]));
      }
    }
  }
  __syncthreads();

  // ---- phase 2: object MLP over 60 objects (waves 0..3, 16 columns each) ----
  if (wave < 4) {
    v16bf V1[4], V2[4][2], V3[2];
    v8f   D1b[4], D2b[4], D3b;
#pragma unroll
    for (int ct = 0; ct < 4; ++ct) {
      const int cte = ct ^ half;
      V1[ct]    = load_w_afrag(fo1w, 32, cte * 16 + lrow, 0,  half, false);
      V2[ct][0] = load_w_afrag(fo2w, 64, cte * 16 + lrow, 0,  half, true);
      V2[ct][1] = load_w_afrag(fo2w, 64, cte * 16 + lrow, 32, half, true);
      D1b[ct]   = load_bias_c(fo1b, cte * 16, half);
      D2b[ct]   = load_bias_c(fo2b, cte * 16, half);
    }
    V3[0] = load_w_afrag(fo3w, 64, lrow, 0,  half, true);
    V3[1] = load_w_afrag(fo3w, 64, lrow, 32, half, true);
    D3b   = load_bias_c(fo3b, 0, half);

    const int obj = wave * 16 + lrow;
    const int oc  = obj < N_OBJ ? obj : (N_OBJ - 1);

    // B: k<16 -> x features (h=0 lanes), k>=16 -> ebar features (h=1 lanes)
    v8u bu;
    if (half == 0) {
      bu = *(const v8u*)&xbf[oc * P_F];
    } else {
#pragma unroll
      for (int j = 0; j < 8; ++j)
        bu[j] = pack2(ebar[oc * DE_F + 2 * j], ebar[oc * DE_F + 2 * j + 1]);
    }
    v16bf B1f = __builtin_bit_cast(v16bf, bu);

    v8f acc[4];
#pragma unroll
    for (int ct = 0; ct < 4; ++ct) acc[ct] = wmma_bf16(V1[ct], B1f, D1b[ct]);

    v16bf B2f0, B2f1;
    next_bfrags(acc, B2f0, B2f1);
#pragma unroll
    for (int ct = 0; ct < 4; ++ct) {
      acc[ct] = wmma_bf16(V2[ct][0], B2f0, D2b[ct]);
      acc[ct] = wmma_bf16(V2[ct][1], B2f1, acc[ct]);
    }

    v16bf B3f0, B3f1;
    next_bfrags(acc, B3f0, B3f1);
    v8f o3 = wmma_bf16(V3[0], B3f0, D3b);
    o3     = wmma_bf16(V3[1], B3f1, o3);

    // O[b, do, n]: lane column = object, feature do = v + 8*half
    if (obj < N_OBJ) {
#pragma unroll
      for (int v = 0; v < 8; ++v) {
        out[(size_t)b * (DO_F * N_OBJ) + (v + 8 * half) * N_OBJ + obj] = reluf(o3[v]);
      }
    }
  }
}

extern "C" void kernel_launch(void* const* d_in, const int* in_sizes, int n_in,
                              void* d_out, int out_size, void* d_ws, size_t ws_size,
                              hipStream_t stream) {
  (void)n_in; (void)out_size; (void)d_ws; (void)ws_size;
  const float* x    = (const float*)d_in[0];
  const float* fr1w = (const float*)d_in[1];
  const float* fr1b = (const float*)d_in[2];
  const float* fr2w = (const float*)d_in[3];
  const float* fr2b = (const float*)d_in[4];
  const float* fr3w = (const float*)d_in[5];
  const float* fr3b = (const float*)d_in[6];
  const float* fo1w = (const float*)d_in[7];
  const float* fo1b = (const float*)d_in[8];
  const float* fo2w = (const float*)d_in[9];
  const float* fo2b = (const float*)d_in[10];
  const float* fo3w = (const float*)d_in[11];
  const float* fo3b = (const float*)d_in[12];
  float* out = (float*)d_out;

  const int batch = in_sizes[0] / (P_F * N_OBJ);   // 512
  in_fused_kernel<<<dim3(batch), dim3(256), 0, stream>>>(
      x, fr1w, fr1b, fr2w, fr2b, fr3w, fr3b,
      fo1w, fo1b, fo2w, fo2b, fo3w, fo3b, out);
}
